// T5Attention_18863496364348
// MI455X (gfx1250) — compile-verified, matrix operand order checked by reading, not run on hardware
//
#include <hip/hip_runtime.h>
#include <hip/hip_bf16.h>
#include <math.h>

// ---------------------------------------------------------------------------
// T5 attention (B=4, H=16, S=1024, D=1024, hd=64) for gfx1250 (wave32, WMMA)
//   0) f32 -> f16 conversion of x and the four weight matrices (workspace)
//   1) Q/K/V projections: block-cooperative WMMA GEMM; weight tiles streamed
//      into double-buffered LDS via global_load_async_to_lds_b128 (ASYNCcnt)
//   2) flash-attention per (b,h,16-query tile): QK^T WMMA -> analytic T5
//      bias + causal mask -> online softmax -> P staged f16 through LDS
//      (s_wait_dscnt) -> P·V WMMA
//   3) output projection WMMA -> f32 d_out
// ---------------------------------------------------------------------------

typedef __attribute__((ext_vector_type(16))) _Float16 v16h;
typedef __attribute__((ext_vector_type(8)))  _Float16 v8h;
typedef __attribute__((ext_vector_type(4)))  _Float16 v4h;
typedef __attribute__((ext_vector_type(8)))  float    v8f;
typedef __attribute__((ext_vector_type(4)))  int      v4i;
typedef __attribute__((address_space(1)))    v4i*     gv4i_p;   // global v4i*
typedef __attribute__((address_space(3)))    v4i*     lv4i_p;   // LDS v4i*

#define BATCH 4
#define HEADS 16
#define SEQ   1024
#define DMODEL 1024
#define HDIM  64
#define NROWS (BATCH * SEQ)          // 4096
#define SCALE_BIAS 0.125f            // 64^-0.5

#if defined(__has_builtin)
#if __has_builtin(__builtin_amdgcn_global_load_async_to_lds_b128)
#define USE_ASYNC_LDS 1
#endif
#endif

// Copy 16 contiguous bytes global -> LDS. Async (ASYNCcnt) when available.
__device__ __forceinline__ void stage16(const _Float16* g, _Float16* l) {
#ifdef USE_ASYNC_LDS
    __builtin_amdgcn_global_load_async_to_lds_b128(
        (gv4i_p)(uintptr_t)g,
        (lv4i_p)(unsigned)(uintptr_t)l,
        /*offset=*/0, /*cpol=*/0);
#else
    *(v8h*)l = *(const v8h*)g;
#endif
}

// Build a 16x16x32 f16 WMMA A/B operand for this lane.
// p points at the 32-element K-chunk of this lane's row (A: row m = lane%16,
// B: row n = lane%16 of the [N][K] matrix). half = lane/16.
// Element e maps to k = (e>=8 ? 16 : 0) + half*8 + (e%8)  (ISA 7.12.2).
__device__ __forceinline__ v16h load_op16(const _Float16* p, int half) {
    const v8h x = *(const v8h*)(p + half * 8);
    const v8h y = *(const v8h*)(p + 16 + half * 8);
    v16h r;
#pragma unroll
    for (int i = 0; i < 8; ++i) { r[i] = x[i]; r[i + 8] = y[i]; }
    return r;
}

__device__ __forceinline__ v8f wmma16(v16h a, v16h b, v8f c) {
    return __builtin_amdgcn_wmma_f32_16x16x32_f16(
        /*neg_a=*/false, a, /*neg_b=*/false, b,
        /*c_mod=*/(short)0, c, /*reuse_a=*/false, /*reuse_b=*/false);
}

// T5 causal relative-position bias (rel = qi - kj >= 0)
__device__ __forceinline__ float t5_bias(int rel, int h, const float* __restrict__ rel_emb) {
    int bucket;
    if (rel < 16) {
        bucket = rel;
    } else {
        // 16 + log(rel/16)/log(8) * 16, truncated, capped at 31
        int v = 16 + (int)(logf((float)rel * 0.0625f) * 7.6943724f);
        bucket = v < 31 ? v : 31;
    }
    return rel_emb[bucket * HEADS + h] * SCALE_BIAS;
}

__global__ __launch_bounds__(256) void cvt_f32_to_f16(const float* __restrict__ in,
                                                      _Float16* __restrict__ out, int n4) {
    int i = blockIdx.x * 256 + threadIdx.x;
    if (i < n4) {
        float4 f = ((const float4*)in)[i];
        v4h o; o[0] = (_Float16)f.x; o[1] = (_Float16)f.y;
               o[2] = (_Float16)f.z; o[3] = (_Float16)f.w;
        ((v4h*)out)[i] = o;
    }
}

// C[row, col] = A[row, :] . W[col, :] + bias[col]     (torch Linear x @ W^T + b)
// A: [4096 x 1024] f16, W: [1024 x 1024] f16.
// Block = 8 waves = one 128(M) x 64(N) tile; each wave owns 16 rows.
// The 64x32 weight tile per k-step is staged into double-buffered LDS with
// async-to-LDS loads so the k+1 prefetch overlaps the k compute.
// mode 0: store f16 [b,h,s,d] (Q)   mode 1: same (K)
// mode 2: store f16 [b,h,d,s] (V^T) mode 3: store f32 row-major (final out)
__global__ __launch_bounds__(256) void gemm_wmma(const _Float16* __restrict__ A,
                                                 const _Float16* __restrict__ W,
                                                 const float* __restrict__ bias,
                                                 void* __restrict__ outp, int mode) {
    __shared__ _Float16 wtile[2][64][32];        // 2 x 4 KB double buffer

    const int tid  = threadIdx.x;
    const int lane = tid & 31;
    const int wid  = tid >> 5;
    const int bm = blockIdx.x >> 4;              // 0..31 (row block of 128)
    const int bn = blockIdx.x & 15;              // 0..15 (64-col strip)
    const int mt = bm * 8 + wid;                 // 16-row tile of this wave
    const int n  = lane & 15;
    const int half = lane >> 4;

    const _Float16* arow = A + (size_t)(mt * 16 + n) * DMODEL;   // A row m = lane%16
    __builtin_prefetch(arow, 0, 3);

    // cooperative staging: 256 threads x 16B = 64x32 f16 tile
    const int srow = tid >> 2;                   // 0..63
    const int scol = (tid & 3) * 8;              // 0,8,16,24
    const _Float16* wsrc = W + (size_t)(bn * 64 + srow) * DMODEL + scol;

    v8f acc[4] = {};
    const int KSTEPS = DMODEL / 32;              // 32

    stage16(wsrc, &wtile[0][srow][scol]);        // prefetch k-chunk 0

    for (int kc = 0; kc < KSTEPS; ++kc) {
        const int cur = kc & 1;
        if (kc + 1 < KSTEPS) {
            stage16(wsrc + (kc + 1) * 32, &wtile[cur ^ 1][srow][scol]);
            asm volatile("s_wait_asynccnt 0x1" ::: "memory");   // cur buffer done
        } else {
            asm volatile("s_wait_asynccnt 0x0" ::: "memory");
        }
        __syncthreads();                         // all waves' pieces visible

        const v16h a  = load_op16(arow + kc * 32, half);
        const v16h b0 = load_op16(&wtile[cur][ 0 + n][0], half);
        const v16h b1 = load_op16(&wtile[cur][16 + n][0], half);
        const v16h b2 = load_op16(&wtile[cur][32 + n][0], half);
        const v16h b3 = load_op16(&wtile[cur][48 + n][0], half);
        acc[0] = wmma16(a, b0, acc[0]);
        acc[1] = wmma16(a, b1, acc[1]);
        acc[2] = wmma16(a, b2, acc[2]);
        acc[3] = wmma16(a, b3, acc[3]);

        __syncthreads();                         // reads done before overwrite
    }

#pragma unroll
    for (int t = 0; t < 4; ++t) {
        const int col = bn * 64 + t * 16 + n;
        const float bb = bias[col];
#pragma unroll
        for (int r = 0; r < 8; ++r) {
            const int row = mt * 16 + r + 8 * half;   // C layout: m = r + 8*(lane/16)
            const float val = acc[t][r] + bb;
            if (mode == 3) {
                ((float*)outp)[(size_t)row * DMODEL + col] = val;
            } else {
                const int b = row >> 10, s = row & 1023;
                const int h = col >> 6,  d = col & 63;
                _Float16* o16 = (_Float16*)outp;
                if (mode == 2)  // V transposed: [b,h,d,s]
                    o16[((size_t)(b * HEADS + h) * HDIM + d) * SEQ + s] = (_Float16)val;
                else            // Q/K: [b,h,s,d]
                    o16[((size_t)(b * HEADS + h) * SEQ + s) * HDIM + d] = (_Float16)val;
            }
        }
    }
}

// Flash attention: one wave per (b, h, 16-query tile). 32-key chunks.
__global__ __launch_bounds__(256) void attn_wmma(const _Float16* __restrict__ Qh,
                                                 const _Float16* __restrict__ Kh,
                                                 const _Float16* __restrict__ Vt,
                                                 const float* __restrict__ rel_emb,
                                                 _Float16* __restrict__ Oh) {
    __shared__ _Float16 pstage[8][16][32];      // 1 KB per wave (P staging)

    const int lane = threadIdx.x & 31;
    const int wid  = threadIdx.x >> 5;
    const int gw   = blockIdx.x * 8 + wid;      // 0..4095
    const int b  = gw >> 10;
    const int h  = (gw >> 6) & 15;
    const int qt = gw & 63;
    const int n    = lane & 15;
    const int half = lane >> 4;

    const _Float16* qbase = Qh + (size_t)(b * HEADS + h) * SEQ * HDIM;
    const _Float16* kbase = Kh + (size_t)(b * HEADS + h) * SEQ * HDIM;
    const _Float16* vbase = Vt + (size_t)(b * HEADS + h) * HDIM * SEQ;

    // Q tile resident: A-operand row m = lane%16, two 32-wide K chunks of hd
    const _Float16* qrow = qbase + (size_t)(qt * 16 + n) * HDIM;
    const v16h qa0 = load_op16(qrow, half);
    const v16h qa1 = load_op16(qrow + 32, half);

    v8f o0 = {}, o1 = {}, o2 = {}, o3 = {};
    float mrun[8], lrun[8];
#pragma unroll
    for (int r = 0; r < 8; ++r) { mrun[r] = -1e30f; lrun[r] = 0.f; }

    const int nk = qt * 16 + 16;                // keys 0 .. qt*16+15
    const int nchunks = (nk + 31) >> 5;

    for (int kc = 0; kc < nchunks; ++kc) {
        const int kb = kc * 32;

        // ---- scores S = Q K^T (two 16x16 tiles covering 32 keys) ----
        const _Float16* krow0 = kbase + (size_t)(kb + n) * HDIM;
        const _Float16* krow1 = kbase + (size_t)(kb + 16 + n) * HDIM;
        v8f s0 = {}, s1 = {};
        s0 = wmma16(qa0, load_op16(krow0, half),      s0);
        s0 = wmma16(qa1, load_op16(krow0 + 32, half), s0);
        s1 = wmma16(qa0, load_op16(krow1, half),      s1);
        s1 = wmma16(qa1, load_op16(krow1 + 32, half), s1);

        // ---- bias + causal mask + online softmax ----
#pragma unroll
        for (int r = 0; r < 8; ++r) {
            const int qi = qt * 16 + r + 8 * half;
            const int rel0 = qi - (kb + n);
            const int rel1 = qi - (kb + 16 + n);
            float v0 = (rel0 < 0) ? -1e30f : s0[r] + t5_bias(rel0, h, rel_emb);
            float v1 = (rel1 < 0) ? -1e30f : s1[r] + t5_bias(rel1, h, rel_emb);

            float mx = fmaxf(v0, v1);
            mx = fmaxf(mx, __shfl_xor(mx, 1, 32));
            mx = fmaxf(mx, __shfl_xor(mx, 2, 32));
            mx = fmaxf(mx, __shfl_xor(mx, 4, 32));
            mx = fmaxf(mx, __shfl_xor(mx, 8, 32));

            const float mn = fmaxf(mrun[r], mx);
            const float scale = expf(mrun[r] - mn);
            const float p0 = expf(v0 - mn);
            const float p1 = expf(v1 - mn);
            float rs = p0 + p1;
            rs += __shfl_xor(rs, 1, 32);
            rs += __shfl_xor(rs, 2, 32);
            rs += __shfl_xor(rs, 4, 32);
            rs += __shfl_xor(rs, 8, 32);

            lrun[r] = lrun[r] * scale + rs;
            mrun[r] = mn;
            o0[r] *= scale; o1[r] *= scale; o2[r] *= scale; o3[r] *= scale;

            const int mrow = r + 8 * half;      // C-layout row
            pstage[wid][mrow][n]      = (_Float16)p0;
            pstage[wid][mrow][16 + n] = (_Float16)p1;
        }

        // wave-private LDS region; only need DS-counter ordering, no barrier
        asm volatile("s_wait_dscnt 0x0" ::: "memory");

        // ---- O += P V : P as A-operand (16x32), V^T chunks as B-operands ----
        const v16h pa = load_op16(&pstage[wid][n][0], half);
        o0 = wmma16(pa, load_op16(vbase + (size_t)(0 * 16 + n) * SEQ + kb, half), o0);
        o1 = wmma16(pa, load_op16(vbase + (size_t)(1 * 16 + n) * SEQ + kb, half), o1);
        o2 = wmma16(pa, load_op16(vbase + (size_t)(2 * 16 + n) * SEQ + kb, half), o2);
        o3 = wmma16(pa, load_op16(vbase + (size_t)(3 * 16 + n) * SEQ + kb, half), o3);
    }

    // ---- epilogue: normalize, store f16 [b, s, h*64+d] for the out-proj ----
#pragma unroll
    for (int r = 0; r < 8; ++r) {
        const float inv = 1.0f / lrun[r];
        const int s = qt * 16 + r + 8 * half;
        _Float16* orow = Oh + (size_t)(b * SEQ + s) * DMODEL + h * HDIM;
        orow[ 0 + n] = (_Float16)(o0[r] * inv);
        orow[16 + n] = (_Float16)(o1[r] * inv);
        orow[32 + n] = (_Float16)(o2[r] * inv);
        orow[48 + n] = (_Float16)(o3[r] * inv);
    }
}

extern "C" void kernel_launch(void* const* d_in, const int* in_sizes, int n_in,
                              void* d_out, int out_size, void* d_ws, size_t ws_size,
                              hipStream_t stream) {
    const float* x    = (const float*)d_in[0];
    const float* Wq   = (const float*)d_in[1];
    const float* bq   = (const float*)d_in[2];
    const float* Wk   = (const float*)d_in[3];
    const float* bk   = (const float*)d_in[4];
    const float* Wv   = (const float*)d_in[5];
    const float* bv   = (const float*)d_in[6];
    const float* Wo   = (const float*)d_in[7];
    const float* bo   = (const float*)d_in[8];
    const float* rel  = (const float*)d_in[9];

    // workspace carve-up (f16 elements); total ~50.3 MB
    const size_t NX = (size_t)NROWS * DMODEL;        // 4096*1024
    const size_t NW = (size_t)DMODEL * DMODEL;       // 1024*1024
    _Float16* xh  = (_Float16*)d_ws;
    _Float16* Wqh = xh  + NX;
    _Float16* Wkh = Wqh + NW;
    _Float16* Wvh = Wkh + NW;
    _Float16* Woh = Wvh + NW;
    _Float16* Qh  = Woh + NW;
    _Float16* Kh  = Qh  + NX;
    _Float16* Vt  = Kh  + NX;
    _Float16* Oh  = Vt  + NX;

    // 0) f32 -> f16 (vectorized: 4 elements / thread)
    cvt_f32_to_f16<<<(int)(NX / 4 / 256), 256, 0, stream>>>(x,  xh,  (int)(NX / 4));
    cvt_f32_to_f16<<<(int)(NW / 4 / 256), 256, 0, stream>>>(Wq, Wqh, (int)(NW / 4));
    cvt_f32_to_f16<<<(int)(NW / 4 / 256), 256, 0, stream>>>(Wk, Wkh, (int)(NW / 4));
    cvt_f32_to_f16<<<(int)(NW / 4 / 256), 256, 0, stream>>>(Wv, Wvh, (int)(NW / 4));
    cvt_f32_to_f16<<<(int)(NW / 4 / 256), 256, 0, stream>>>(Wo, Woh, (int)(NW / 4));

    // 1) projections: 32 row-blocks x 16 col-strips = 512 blocks of 8 waves
    gemm_wmma<<<512, 256, 0, stream>>>(xh, Wqh, bq, Qh, 0);
    gemm_wmma<<<512, 256, 0, stream>>>(xh, Wkh, bk, Kh, 1);
    gemm_wmma<<<512, 256, 0, stream>>>(xh, Wvh, bv, Vt, 2);

    // 2) flash attention: 4096 waves (b*h*qtile) / 8 per block
    attn_wmma<<<512, 256, 0, stream>>>(Qh, Kh, Vt, rel, Oh);

    // 3) output projection -> f32 d_out
    gemm_wmma<<<512, 256, 0, stream>>>(Oh, Woh, bo, d_out, 3);
}
